// Qwen3MoeAttention_53042846105768
// MI455X (gfx1250) — matrix operand order repaired and checked by reading, not compiled, not run on hardware
//
#include <hip/hip_runtime.h>
#include <hip/hip_bf16.h>
#include <cstdint>

// ---------------------------------------------------------------------------
// Qwen3-MoE attention layer for gfx1250 (MI455X), wave32 + WMMA + async-LDS.
//   T=2048, HID=2048, H=16, KV=4, HD=128, theta=1e6
// Pipeline:
//   1) cvt hidden -> f16 [T][K]
//   2) transpose+cvt w_qkv -> f16 [N][K], w_o -> f16 [N][K]
//   3) WMMA GEMM (async global->LDS, ping-pong, compile-time buffer index):
//      qkv_f32 = X @ Wqkv
//   4) RoPE + extract: q_f16[H][T][HD], k_f16[KV][T][HD], v_t[KV][HD][T]
//   5) flash attention (wave-per-16-rows, online softmax) -> o_f16 [T][HID]
//   6) WMMA GEMM: out_f32 = O @ Wo
// ---------------------------------------------------------------------------

#define T_SEQ 2048
#define HIDN  2048
#define NH    16
#define NKV   4
#define HD    128
#define QSIZE  (NH * HD)                 // 2048
#define KVSIZE (NKV * HD)                // 512
#define QKVN   (QSIZE + 2 * KVSIZE)      // 3072

typedef __attribute__((ext_vector_type(16))) _Float16 v16h;
typedef __attribute__((ext_vector_type(8)))  float    v8f;

__device__ __forceinline__ v8f zero8() {
    v8f z = {0.f, 0.f, 0.f, 0.f, 0.f, 0.f, 0.f, 0.f};
    return z;
}

__device__ __forceinline__ v8f wmma_f16(v16h a, v16h b, v8f c) {
    // D = A(16x32 f16) * B(32x16 f16) + C(16x16 f32)
    return __builtin_amdgcn_wmma_f32_16x16x32_f16(
        false, a, false, b, (short)0, c, false, false);
}

// Generic pointer to __shared__ is {aperture_hi32, lds_byte_offset} per the
// flat-aperture rules, so the low 32 bits are the LDS address the async
// instructions need in VDST.
__device__ __forceinline__ uint32_t lds_off32(const void* p) {
    return (uint32_t)(uintptr_t)p;
}

// GLOBAL_LOAD_ASYNC_TO_LDS_B128, GVS mode: 64-bit SGPR base + 32-bit VGPR
// byte offset; VDST VGPR holds destination LDS byte address. ASYNCcnt-tracked.
#define ASYNC_LD_B128(ldsoff, goff, base)                         \
    asm volatile("global_load_async_to_lds_b128 %0, %1, %2"       \
                 :: "v"(ldsoff), "v"(goff), "s"(base) : "memory")

#define WAIT_ASYNC(n) asm volatile("s_wait_asynccnt " #n ::: "memory")

// A operand: 16x32 f16 from row-major [row][K] storage.
// lane<16: row=lane, K chunks [0..7] and [16..23]; lane>=16: row=lane-16,
// K chunks [8..15] and [24..31].
__device__ __forceinline__ v16h load_frag_a(const _Float16* base, int stride) {
    int lane = threadIdx.x & 31;
    int row  = lane & 15;
    int hi   = lane >> 4;
    const _Float16* p = base + row * stride + hi * 8;
    v16h f;
    *(float4*)&f       = *(const float4*)p;         // 8 halfs
    *((float4*)&f + 1) = *(const float4*)(p + 16);  // 8 halfs
    return f;
}

// B operand: 32x16 f16 where storage is row-major [n][K] (pre-transposed).
// lane<16: col n=lane, K [0..7] and [16..23]; lane>=16: n=lane-16, K [8..15] and [24..31].
__device__ __forceinline__ v16h load_frag_b(const _Float16* base, int stride) {
    int lane = threadIdx.x & 31;
    int row  = lane & 15;
    int hi   = lane >> 4;
    const _Float16* p = base + row * stride + hi * 8;
    v16h f;
    *(float4*)&f       = *(const float4*)p;
    *((float4*)&f + 1) = *(const float4*)(p + 16);
    return f;
}

// ---------------------------------------------------------------------------
// elementwise f32 -> f16
__global__ __launch_bounds__(256) void cvt_f16_kernel(
    const float* __restrict__ in, _Float16* __restrict__ out, int n) {
    int i = blockIdx.x * 256 + threadIdx.x;
    if (i < n) out[i] = (_Float16)in[i];
}

// f32 [R][C] -> f16 [C][R]
__global__ __launch_bounds__(256) void transpose_cvt_kernel(
    const float* __restrict__ in, _Float16* __restrict__ out, int R, int C) {
    int i = blockIdx.x * 256 + threadIdx.x;
    if (i < R * C) {
        int r = i / C, c = i % C;
        out[(size_t)c * R + r] = (_Float16)in[i];
    }
}

// ---------------------------------------------------------------------------
// C[M][N] (f32) = A[M][K] (f16, row-major) * Bt[N][K] (f16, row-major = B^T)
// 128x128 block tile, BK=32, 8 waves each computing 32x64.
// Tiles staged with ASYNC global->LDS copies; K-loop unrolled by 2 so the
// ping-pong buffer index is a compile-time constant (no cndmask selects).
#define LDSTR 40   // 32 + 8 halfs pad: rows 80B apart (16B aligned, bank-spread)

__global__ __launch_bounds__(256) void wmma_gemm_f16_kernel(
    const _Float16* __restrict__ A, const _Float16* __restrict__ Bt,
    float* __restrict__ C, int M, int N, int K) {
    __shared__ _Float16 As[2][128 * LDSTR];
    __shared__ _Float16 Bs[2][128 * LDSTR];

    int tid  = threadIdx.x;
    int wave = tid >> 5;
    int lane = tid & 31;
    int bm = blockIdx.y * 128;
    int bn = blockIdx.x * 128;
    int wm = (wave >> 1) * 32;  // wave row offset in tile
    int wn = (wave & 1) * 64;   // wave col offset in tile

    uint64_t abase = (uint64_t)(uintptr_t)A;
    uint64_t bbase = (uint64_t)(uintptr_t)Bt;

    // Per-thread staging slots: 2 x 16B chunks of each 128x32 tile.
    uint32_t aoff[2], boff[2], ldsA[2][2], ldsB[2][2];
#pragma unroll
    for (int r = 0; r < 2; ++r) {
        int c    = tid * 2 + r;   // 0..511
        int row  = c >> 2;
        int col8 = (c & 3) * 8;
        aoff[r] = (uint32_t)(((bm + row) * K + col8) * 2);  // bytes, + kt*2
        boff[r] = (uint32_t)(((bn + row) * K + col8) * 2);
#pragma unroll
        for (int buf = 0; buf < 2; ++buf) {
            ldsA[buf][r] = lds_off32(&As[buf][row * LDSTR + col8]);
            ldsB[buf][r] = lds_off32(&Bs[buf][row * LDSTR + col8]);
        }
    }

    v8f acc[2][4];
#pragma unroll
    for (int i = 0; i < 2; ++i)
#pragma unroll
        for (int j = 0; j < 4; ++j) acc[i][j] = zero8();

    // Issue the 4 async copies for K-chunk at byte offset kt*2 into `buf`.
    auto stage = [&](int buf, int kt) __attribute__((always_inline)) {
        uint32_t ktb = (uint32_t)(kt * 2);
#pragma unroll
        for (int r = 0; r < 2; ++r) {
            ASYNC_LD_B128(ldsA[buf][r], aoff[r] + ktb, abase);
            ASYNC_LD_B128(ldsB[buf][r], boff[r] + ktb, bbase);
        }
    };
    // 8 WMMAs on the staged 128x32 tiles in `buf`.
    auto compute = [&](int buf) __attribute__((always_inline)) {
        v16h af[2], bf[4];
#pragma unroll
        for (int mi = 0; mi < 2; ++mi)
            af[mi] = load_frag_a(&As[buf][(wm + mi * 16) * LDSTR], LDSTR);
#pragma unroll
        for (int ni = 0; ni < 4; ++ni)
            bf[ni] = load_frag_b(&Bs[buf][(wn + ni * 16) * LDSTR], LDSTR);
#pragma unroll
        for (int mi = 0; mi < 2; ++mi)
#pragma unroll
            for (int ni = 0; ni < 4; ++ni)
                acc[mi][ni] = wmma_f16(af[mi], bf[ni], acc[mi][ni]);
    };

    // prologue: stage K-chunk 0 into buffer 0
    stage(0, 0);

    // steady state: two K-chunks per trip, unconditional prefetch + wait<=4
    int kt = 0;
    for (; kt + 64 < K; kt += 64) {
        __syncthreads();            // buf1 free (prev compute(1) done)
        stage(1, kt + 32);
        WAIT_ASYNC(0x4);            // in-order: buf0 chunk has landed
        __syncthreads();            // ... in every wave
        compute(0);
        __syncthreads();            // buf0 free
        stage(0, kt + 64);
        WAIT_ASYNC(0x4);            // buf1 chunk has landed
        __syncthreads();
        compute(1);
    }
    // tail: last two chunks (kt == K-64)
    __syncthreads();
    stage(1, K - 32);
    WAIT_ASYNC(0x4);
    __syncthreads();
    compute(0);
    WAIT_ASYNC(0x0);                // own buf1 copies landed
    __syncthreads();                // everyone's landed
    compute(1);

    int hi  = lane >> 4;
    int col = lane & 15;
#pragma unroll
    for (int mi = 0; mi < 2; ++mi)
#pragma unroll
        for (int ni = 0; ni < 4; ++ni)
#pragma unroll
            for (int v = 0; v < 8; ++v) {
                int r  = bm + wm + mi * 16 + v + hi * 8;
                int cn = bn + wn + ni * 16 + col;
                C[(size_t)r * N + cn] = acc[mi][ni][v];
            }
}

// ---------------------------------------------------------------------------
// RoPE + layout extraction from qkv f32 [T][3072].
__global__ __launch_bounds__(256) void rope_extract_kernel(
    const float* __restrict__ qkv, _Float16* __restrict__ qf,
    _Float16* __restrict__ kf, _Float16* __restrict__ vt,
    const int* __restrict__ positions) {
    int t   = blockIdx.x;
    int tid = threadIdx.x;
    float pos = (float)positions[t];
    const float* row = qkv + (size_t)t * QKVN;

    int j = tid & 63;                 // rotary pair index, half = 64
    float inv = __powf(1.0e6f, -(float)j * (1.0f / 64.0f));
    float fr  = pos * inv;
    float cs  = __cosf(fr);
    float sn  = __sinf(fr);

    // q heads, RoPE: q_f16 [H][T][HD]
    for (int h = tid >> 6; h < NH; h += 4) {
        float x1 = row[h * HD + j];
        float x2 = row[h * HD + 64 + j];
        size_t o = ((size_t)h * T_SEQ + t) * HD;
        qf[o + j]      = (_Float16)(x1 * cs - x2 * sn);
        qf[o + 64 + j] = (_Float16)(x2 * cs + x1 * sn);
    }
    // k heads, RoPE: k_f16 [KV][T][HD]
    {
        int h = tid >> 6;
        if (h < NKV) {
            float x1 = row[QSIZE + h * HD + j];
            float x2 = row[QSIZE + h * HD + 64 + j];
            size_t o = ((size_t)h * T_SEQ + t) * HD;
            kf[o + j]      = (_Float16)(x1 * cs - x2 * sn);
            kf[o + 64 + j] = (_Float16)(x2 * cs + x1 * sn);
        }
    }
    // v transposed: v_t [KV][HD][T]
    for (int i = tid; i < NKV * HD; i += 256) {
        int h = i >> 7, d = i & 127;
        vt[((size_t)h * HD + d) * T_SEQ + t] =
            (_Float16)row[QSIZE + KVSIZE + i];
    }
}

// ---------------------------------------------------------------------------
// Flash attention. grid = (T/128, H), 8 waves; wave owns 16 query rows.
__global__ __launch_bounds__(256) void flash_attn_kernel(
    const _Float16* __restrict__ qf, const _Float16* __restrict__ kf,
    const _Float16* __restrict__ vt, _Float16* __restrict__ of) {
    __shared__ _Float16 ptile[8][16 * LDSTR];  // per-wave 16x32 P (padded)

    int h    = blockIdx.y;
    int kvh  = h >> 2;                 // GQA group = 4
    int wave = threadIdx.x >> 5;
    int lane = threadIdx.x & 31;
    int t0   = blockIdx.x * 128 + wave * 16;
    int col  = lane & 15;
    int hi   = lane >> 4;
    const float scale = 0.08838834764831845f;  // 1/sqrt(128)

    // Q operand fragments for 16 rows x HD=128 (4 K-chunks of 32)
    const _Float16* qbase = qf + ((size_t)h * T_SEQ + t0) * HD;
    v16h qfr[4];
#pragma unroll
    for (int c = 0; c < 4; ++c) qfr[c] = load_frag_a(qbase + c * 32, HD);

    v8f acc[8];
#pragma unroll
    for (int d = 0; d < 8; ++d) acc[d] = zero8();
    float m[8], l[8];
#pragma unroll
    for (int v = 0; v < 8; ++v) { m[v] = -3.0e38f; l[v] = 0.f; }

    int tmax = t0 + 15;
    int nsb  = (tmax >> 5) + 1;        // key blocks of 32, causal bound
    for (int sb = 0; sb < nsb; ++sb) {
        int sbase = sb * 32;
        const _Float16* kbase = kf + ((size_t)kvh * T_SEQ + sbase) * HD;
        const _Float16* vb = vt + (size_t)kvh * HD * T_SEQ + sbase;

        if (sb + 1 < nsb) {  // prefetch next key/value block into caches
            __builtin_prefetch(kbase + 32 * HD, 0, 0);
            __builtin_prefetch(vb + 32, 0, 0);
        }

        // S = Q K^T for 16 rows x 32 keys (two 16x16 accumulators)
        v8f s0 = zero8(), s1 = zero8();
#pragma unroll
        for (int c = 0; c < 4; ++c) {
            v16h k0 = load_frag_b(kbase + c * 32, HD);
            v16h k1 = load_frag_b(kbase + 16 * HD + c * 32, HD);
            s0 = wmma_f16(qfr[c], k0, s0);
            s1 = wmma_f16(qfr[c], k1, s1);
        }

        // online softmax; row r of a C tile lives in one 16-lane half
#pragma unroll
        for (int v = 0; v < 8; ++v) {
            int trow = t0 + v + hi * 8;
            float a0 = s0[v] * scale;
            float a1 = s1[v] * scale;
            if (sbase + col > trow)      a0 = -1.0e30f;
            if (sbase + 16 + col > trow) a1 = -1.0e30f;
            float mx = fmaxf(a0, a1);
            mx = fmaxf(mx, __shfl_xor(mx, 1));
            mx = fmaxf(mx, __shfl_xor(mx, 2));
            mx = fmaxf(mx, __shfl_xor(mx, 4));
            mx = fmaxf(mx, __shfl_xor(mx, 8));
            float mn  = fmaxf(m[v], mx);
            float rsc = __expf(m[v] - mn);
            float p0  = __expf(a0 - mn);
            float p1  = __expf(a1 - mn);
            float rs  = p0 + p1;
            rs += __shfl_xor(rs, 1);
            rs += __shfl_xor(rs, 2);
            rs += __shfl_xor(rs, 4);
            rs += __shfl_xor(rs, 8);
            l[v] = l[v] * rsc + rs;
            m[v] = mn;
#pragma unroll
            for (int d = 0; d < 8; ++d) acc[d][v] *= rsc;
            int prow = v + hi * 8;
            ptile[wave][prow * LDSTR + col]      = (_Float16)p0;
            ptile[wave][prow * LDSTR + 16 + col] = (_Float16)p1;
        }

        // P (16x32) back through wave-private LDS to A-fragment layout
        v16h pf = load_frag_a(&ptile[wave][0], LDSTR);

        // acc += P * V  (V_t rows are head-dim, cols are keys: [n][k])
#pragma unroll
        for (int d = 0; d < 8; ++d) {
            v16h vfrag = load_frag_b(vb + (size_t)(d * 16) * T_SEQ, T_SEQ);
            acc[d] = wmma_f16(pf, vfrag, acc[d]);
        }
    }

    // epilogue: o_f16 [T][HID], normalized
#pragma unroll
    for (int d = 0; d < 8; ++d)
#pragma unroll
        for (int v = 0; v < 8; ++v) {
            int trow = t0 + v + hi * 8;
            of[(size_t)trow * HIDN + h * HD + d * 16 + col] =
                (_Float16)(acc[d][v] / l[v]);
        }
}

// ---------------------------------------------------------------------------
extern "C" void kernel_launch(void* const* d_in, const int* in_sizes, int n_in,
                              void* d_out, int out_size, void* d_ws, size_t ws_size,
                              hipStream_t stream) {
    (void)in_sizes; (void)n_in; (void)out_size; (void)ws_size;
    const float* hs     = (const float*)d_in[0];   // [T][HID]
    const float* w_qkv  = (const float*)d_in[1];   // [HID][QKVN]
    const float* w_o    = (const float*)d_in[2];   // [HID][HID]
    const int*   pos    = (const int*)d_in[3];     // [T]
    float*       out    = (float*)d_out;           // [T][HID]

    char* ws = (char*)d_ws;
    _Float16* Xf  = (_Float16*)ws; ws += (size_t)T_SEQ * HIDN * 2;   //  8 MB
    _Float16* Wqt = (_Float16*)ws; ws += (size_t)QKVN * HIDN * 2;    // 12 MB
    _Float16* Wot = (_Float16*)ws; ws += (size_t)HIDN * HIDN * 2;    //  8 MB
    float*    qkv = (float*)ws;    ws += (size_t)T_SEQ * QKVN * 4;   // 24 MB
    _Float16* qf  = (_Float16*)ws; ws += (size_t)NH  * T_SEQ * HD * 2;  // 8 MB
    _Float16* kf  = (_Float16*)ws; ws += (size_t)NKV * T_SEQ * HD * 2;  // 2 MB
    _Float16* vtp = (_Float16*)ws; ws += (size_t)NKV * HD * T_SEQ * 2;  // 2 MB
    _Float16* of  = (_Float16*)ws; ws += (size_t)T_SEQ * HIDN * 2;      // 8 MB

    cvt_f16_kernel<<<(T_SEQ * HIDN) / 256, 256, 0, stream>>>(hs, Xf, T_SEQ * HIDN);
    transpose_cvt_kernel<<<(HIDN * QKVN) / 256, 256, 0, stream>>>(w_qkv, Wqt, HIDN, QKVN);
    transpose_cvt_kernel<<<(HIDN * HIDN) / 256, 256, 0, stream>>>(w_o, Wot, HIDN, HIDN);

    wmma_gemm_f16_kernel<<<dim3(QKVN / 128, T_SEQ / 128), 256, 0, stream>>>(
        Xf, Wqt, qkv, T_SEQ, QKVN, HIDN);

    rope_extract_kernel<<<T_SEQ, 256, 0, stream>>>(qkv, qf, kf, vtp, pos);

    flash_attn_kernel<<<dim3(T_SEQ / 128, NH), 256, 0, stream>>>(qf, kf, vtp, of);

    wmma_gemm_f16_kernel<<<dim3(HIDN / 128, T_SEQ / 128), 256, 0, stream>>>(
        of, Wot, out, T_SEQ, HIDN, HIDN);
}
